// LongContextAttention_58394375356455
// MI455X (gfx1250) — matrix operand order repaired and checked by reading, not compiled, not run on hardware
//
#include <hip/hip_runtime.h>

// Problem constants (B=1, S=2048, H=16, D=128, fp32 in/out).
#define S_LEN 2048
#define NH    16
#define DH    128
#define BQ    64            // q rows per workgroup (4 waves x 16)
#define BK    64            // keys per staged chunk
#define NW    4             // waves per workgroup
#define NCHUNK (S_LEN / BK)
#define KPAD  (DH + 8)      // halfs per K-lds row (272B = 256B row + 16B TDM pad)
#define VPAD  (BK + 8)      // halfs per V-lds row (144B = 128B row + 16B TDM pad)
#define PPAD  (BK + 8)

#if defined(__has_builtin)
#  if __has_builtin(__builtin_amdgcn_tensor_load_to_lds)
#    define HAVE_TDM 1
#  endif
#endif
#ifndef HAVE_TDM
#  define HAVE_TDM 0
#endif
#if __has_include(<hip/amd_detail/amd_gfx1250_TDM.h>)
#  define TDM_SIX 1
#else
#  define TDM_SIX 0
#endif

typedef __attribute__((ext_vector_type(16))) _Float16 v16h;
typedef __attribute__((ext_vector_type(8)))  _Float16 v8h;
typedef __attribute__((ext_vector_type(4)))  _Float16 v4h;
typedef __attribute__((ext_vector_type(8)))  float    v8f;
typedef unsigned int u32x4 __attribute__((ext_vector_type(4)));
typedef int          i32x4 __attribute__((ext_vector_type(4)));
typedef int          i32x8 __attribute__((ext_vector_type(8)));

static __device__ __forceinline__ v16h join8(v8h lo, v8h hi) {
    return __builtin_shufflevector(lo, hi, 0,1,2,3,4,5,6,7,8,9,10,11,12,13,14,15);
}

// Pure-VALU reductions over each 16-lane row via DPP16.
template <int CTRL>
static __device__ __forceinline__ float dpp_maxs(float x) {
    int y = __builtin_amdgcn_update_dpp(0, __float_as_int(x), CTRL, 0xF, 0xF, true);
    return fmaxf(x, __int_as_float(y));
}
template <int CTRL>
static __device__ __forceinline__ float dpp_adds(float x) {
    int y = __builtin_amdgcn_update_dpp(0, __float_as_int(x), CTRL, 0xF, 0xF, true);
    return x + __int_as_float(y);
}
static __device__ __forceinline__ float row16_max(float x) {
    x = dpp_maxs<0xB1>(x); x = dpp_maxs<0x4E>(x);
    x = dpp_maxs<0x141>(x); x = dpp_maxs<0x140>(x);
    return x;
}
static __device__ __forceinline__ float row16_sum(float x) {
    x = dpp_adds<0xB1>(x); x = dpp_adds<0x4E>(x);
    x = dpp_adds<0x141>(x); x = dpp_adds<0x140>(x);
    return x;
}

#if HAVE_TDM
// Issue a 2D TDM load: global (elem=2B, rows of tile0 elems, row stride stride0
// elems) -> LDS at lds_off with pad_amount(+1) DWORDs inserted after each
// 2^(pad_interval+1) DWORDs (recreates the bank-stagger row padding).
static __device__ __forceinline__ void tdm_load_2d(unsigned lds_off,
                                                   unsigned long long gaddr,
                                                   int pad_interval, int pad_amount,
                                                   int tile0, int tile1, int stride0)
{
    u32x4 g0;
    g0[0] = 1u;                                        // count=1 (valid), user mode
    g0[1] = lds_off;                                   // lds_addr
    g0[2] = (unsigned)(gaddr & 0xFFFFFFFFull);         // global_addr[31:0]
    g0[3] = (unsigned)((gaddr >> 32) & 0x01FFFFFFull)  // global_addr[56:32]
          | (2u << 30);                                // type=2 ("image")
    i32x8 g1;
    g1[0] = (1 << 16)                                  // data_size = 2B
          | (1 << 20)                                  // pad_enable
          | (pad_interval << 22) | (pad_amount << 25);
    g1[1] = (tile0 & 0xFFFF) << 16;                    // tensor_dim0 = tile0
    g1[2] = ((tile1 & 0xFFFF) << 16);                  // tensor_dim1 = tile1
    g1[3] = (tile0 << 16);                             // tile_dim0
    g1[4] = tile1;                                     // tile_dim1 (tile_dim2=0)
    g1[5] = stride0;                                   // tensor_dim0_stride[31:0]
    g1[6] = 0;
    g1[7] = 0;
    i32x4 z4 = {0, 0, 0, 0};
#if TDM_SIX
    i32x8 z8 = {0, 0, 0, 0, 0, 0, 0, 0};
    __builtin_amdgcn_tensor_load_to_lds(g0, g1, z4, z4, z8, 0);
#else
    __builtin_amdgcn_tensor_load_to_lds(g0, g1, z4, z4, 0);
#endif
}
#endif

// ---------------- pre-pass: K -> f16 [h][s][d] ----------------
__global__ __launch_bounds__(256)
void cvt_k_kernel(const float* __restrict__ K, _Float16* __restrict__ K16)
{
    size_t i = (size_t)blockIdx.x * 256 + threadIdx.x;   // indexes [s][h][d]
    int d = (int)(i & (DH - 1));
    int h = (int)((i >> 7) & (NH - 1));
    int s = (int)(i >> 11);
    K16[((size_t)h * S_LEN + s) * DH + d] = (_Float16)K[i];
}

// ---------------- pre-pass: V -> f16 transposed [h][d][s] ----------------
__global__ __launch_bounds__(256)
void cvt_vt_kernel(const float* __restrict__ V, _Float16* __restrict__ V16T)
{
    __shared__ _Float16 tile[64][66];
    const int t  = threadIdx.x;
    const int s0 = blockIdx.x * 64;
    const int d0 = blockIdx.y * 64;
    const int h  = blockIdx.z;
    #pragma unroll
    for (int i = 0; i < 16; ++i) {            // coalesced read over d
        int idx = t + i * 256;
        int sl = idx >> 6, dl = idx & 63;
        tile[sl][dl] = (_Float16)V[((size_t)(s0 + sl) * NH + h) * DH + d0 + dl];
    }
    __syncthreads();
    #pragma unroll
    for (int i = 0; i < 16; ++i) {            // coalesced write over s
        int idx = t + i * 256;
        int dl = idx >> 6, sl = idx & 63;
        V16T[((size_t)h * DH + d0 + dl) * S_LEN + s0 + sl] = tile[sl][dl];
    }
}

// ---------------- main TDM flash-attention kernel ----------------
__global__ __launch_bounds__(128)
void fa_f16wmma_tdm_kernel(const float* __restrict__ Q,
                           const _Float16* __restrict__ K16,
                           const _Float16* __restrict__ V16T,
                           float* __restrict__ O)
{
    __shared__ _Float16 Klds[2][BK][KPAD];   // [buf][key][d]
    __shared__ _Float16 Vlds[2][DH][VPAD];   // [buf][d][key]
    __shared__ _Float16 Plds[NW][16][PPAD];

    const int  t      = threadIdx.x;
    const int  w      = t >> 5;
    const int  lane   = t & 31;
    const int  l16    = lane & 15;
    const bool hihalf = (lane >= 16);
    const int  h      = blockIdx.y;
    const int  qrow0  = blockIdx.x * BQ + w * 16;

    const float qscale = 1.4426950408889634f * 0.08838834764831845f; // log2e/sqrt(128)

    // ---- Q A-fragments ----
    v16h aq[4];
    {
        const float* qb = Q + ((size_t)(qrow0 + l16) * NH + h) * DH + (hihalf ? 8 : 0);
        #pragma unroll
        for (int c = 0; c < 4; ++c) {
            float4 lo0 = *(const float4*)(qb + c * 32);
            float4 lo1 = *(const float4*)(qb + c * 32 + 4);
            float4 hi0 = *(const float4*)(qb + c * 32 + 16);
            float4 hi1 = *(const float4*)(qb + c * 32 + 20);
            v16h a;
            a[0]=(_Float16)(lo0.x*qscale); a[1]=(_Float16)(lo0.y*qscale);
            a[2]=(_Float16)(lo0.z*qscale); a[3]=(_Float16)(lo0.w*qscale);
            a[4]=(_Float16)(lo1.x*qscale); a[5]=(_Float16)(lo1.y*qscale);
            a[6]=(_Float16)(lo1.z*qscale); a[7]=(_Float16)(lo1.w*qscale);
            a[8]=(_Float16)(hi0.x*qscale); a[9]=(_Float16)(hi0.y*qscale);
            a[10]=(_Float16)(hi0.z*qscale); a[11]=(_Float16)(hi0.w*qscale);
            a[12]=(_Float16)(hi1.x*qscale); a[13]=(_Float16)(hi1.y*qscale);
            a[14]=(_Float16)(hi1.z*qscale); a[15]=(_Float16)(hi1.w*qscale);
            aq[c] = a;
        }
    }

    v8f ctx[8];
    #pragma unroll
    for (int n = 0; n < 8; ++n) ctx[n] = (v8f){};
    float mrun[8], lrun[8];
    #pragma unroll
    for (int r = 0; r < 8; ++r) { mrun[r] = -3.0e38f; lrun[r] = 0.0f; }

#if HAVE_TDM
    const unsigned long long kbase = (unsigned long long)(size_t)(K16 + (size_t)h * S_LEN * DH) * 1ull;
    const unsigned long long vbase = (unsigned long long)(size_t)(V16T + (size_t)h * DH * S_LEN) * 1ull;
    if (w == 0) {   // prologue: DMA chunk 0 into buf 0
        tdm_load_2d((unsigned)(size_t)&Klds[0][0][0], kbase, 5, 3, DH, BK, DH);
        tdm_load_2d((unsigned)(size_t)&Vlds[0][0][0], vbase, 4, 3, BK, DH, S_LEN);
    }
#endif

    for (int kc = 0; kc < NCHUNK; ++kc) {
        const int buf = kc & 1;
#if HAVE_TDM
        if (w == 0) __builtin_amdgcn_s_wait_tensorcnt(0);
        __syncthreads();                      // buf ready for all; buf^1 free
        if (w == 0 && kc + 1 < NCHUNK) {      // DMA next chunk into buf^1
            const int kn = (kc + 1) * BK;
            tdm_load_2d((unsigned)(size_t)&Klds[buf ^ 1][0][0],
                        kbase + (size_t)kn * DH * 2, 5, 3, DH, BK, DH);
            tdm_load_2d((unsigned)(size_t)&Vlds[buf ^ 1][0][0],
                        vbase + (size_t)kn * 2, 4, 3, BK, DH, S_LEN);
        }
#else
        // fallback staging: plain f16 b128 copies from preconverted buffers
        {
            const int k0 = kc * BK;
            #pragma unroll
            for (int i = 0; i < 8; ++i) {     // K: 1024 x v8h
                int idx = t + i * 128;
                int row = idx >> 4, c8 = (idx & 15) * 8;
                *(v8h*)&Klds[buf][row][c8] =
                    *(const v8h*)(K16 + ((size_t)h * S_LEN + k0 + row) * DH + c8);
            }
            #pragma unroll
            for (int i = 0; i < 4; ++i) {     // V: 512 x v8h
                int idx = t + i * 128;
                int row = idx >> 3, c8 = (idx & 7) * 8;
                *(v8h*)&Vlds[buf][row][c8] =
                    *(const v8h*)(V16T + ((size_t)h * DH + row) * S_LEN + k0 + c8);
            }
        }
        __syncthreads();
#endif

        // ---- S = Q * K^T : 16 WMMA ----
        v8f s[4];
        #pragma unroll
        for (int kt = 0; kt < 4; ++kt) s[kt] = (v8f){};
        #pragma unroll
        for (int c = 0; c < 4; ++c) {
            #pragma unroll
            for (int kt = 0; kt < 4; ++kt) {
                const _Float16* kp = &Klds[buf][kt * 16 + l16][c * 32 + (hihalf ? 8 : 0)];
                v16h kb = join8(*(const v8h*)kp, *(const v8h*)(kp + 16));
                s[kt] = __builtin_amdgcn_wmma_f32_16x16x32_f16(false, aq[c], false, kb,
                                                               (short)0, s[kt], false, false);
            }
        }

        // ---- online softmax ----
        #pragma unroll
        for (int r = 0; r < 8; ++r) {
            float mx = fmaxf(fmaxf(s[0][r], s[1][r]), fmaxf(s[2][r], s[3][r]));
            mx = row16_max(mx);
            float mo = mrun[r];
            float mn = fmaxf(mo, mx);
            float sc = __builtin_amdgcn_exp2f(mo - mn);
            mrun[r]  = mn;
            float e0 = __builtin_amdgcn_exp2f(s[0][r] - mn);
            float e1 = __builtin_amdgcn_exp2f(s[1][r] - mn);
            float e2 = __builtin_amdgcn_exp2f(s[2][r] - mn);
            float e3 = __builtin_amdgcn_exp2f(s[3][r] - mn);
            s[0][r] = e0; s[1][r] = e1; s[2][r] = e2; s[3][r] = e3;
            float rs = row16_sum((e0 + e1) + (e2 + e3));
            lrun[r] = lrun[r] * sc + rs;
            #pragma unroll
            for (int n = 0; n < 8; ++n) ctx[n][r] *= sc;
        }

        // ---- P relayout through per-wave LDS ----
        {
            const int rbase = hihalf ? 8 : 0;
            #pragma unroll
            for (int r = 0; r < 8; ++r) {
                #pragma unroll
                for (int kt = 0; kt < 4; ++kt)
                    Plds[w][rbase + r][l16 + kt * 16] = (_Float16)s[kt][r];
            }
        }
        const _Float16* pp = &Plds[w][l16][hihalf ? 16 : 0];
        v16h pfrag0 = join8(*(const v8h*)pp,        *(const v8h*)(pp + 8));
        v16h pfrag1 = join8(*(const v8h*)(pp + 32), *(const v8h*)(pp + 40));

        // ---- ctx += P * V : 16 WMMA ----
        #pragma unroll
        for (int n = 0; n < 8; ++n) {
            const _Float16* vp = &Vlds[buf][n * 16 + l16][hihalf ? 16 : 0];
            v16h vb0 = join8(*(const v8h*)vp,        *(const v8h*)(vp + 8));
            ctx[n] = __builtin_amdgcn_wmma_f32_16x16x32_f16(false, pfrag0, false, vb0,
                                                            (short)0, ctx[n], false, false);
            v16h vb1 = join8(*(const v8h*)(vp + 32), *(const v8h*)(vp + 40));
            ctx[n] = __builtin_amdgcn_wmma_f32_16x16x32_f16(false, pfrag1, false, vb1,
                                                            (short)0, ctx[n], false, false);
        }
        __syncthreads();
    }

    // ---- normalize and store ----
    #pragma unroll
    for (int r = 0; r < 8; ++r) {
        float inv = 1.0f / lrun[r];
        int row = qrow0 + (hihalf ? 8 : 0) + r;
        float* ob = O + ((size_t)row * NH + h) * DH + l16;
        #pragma unroll
        for (int n = 0; n < 8; ++n) ob[n * 16] = ctx[n][r] * inv;
    }
}

// ---------------- fallback (no-workspace) kernel: round-2 version ----------------
__global__ __launch_bounds__(128)
void fa_f16wmma_kernel(const float* __restrict__ Q, const float* __restrict__ K,
                       const float* __restrict__ V, float* __restrict__ O)
{
    __shared__ _Float16 Klds[BK][KPAD];
    __shared__ _Float16 Vlds[DH][VPAD];
    __shared__ _Float16 Plds[NW][16][PPAD];

    const int  t      = threadIdx.x;
    const int  w      = t >> 5;
    const int  lane   = t & 31;
    const int  l16    = lane & 15;
    const bool hihalf = (lane >= 16);
    const int  h      = blockIdx.y;
    const int  qrow0  = blockIdx.x * BQ + w * 16;
    const float qscale = 1.4426950408889634f * 0.08838834764831845f;

    v16h aq[4];
    {
        const float* qb = Q + ((size_t)(qrow0 + l16) * NH + h) * DH + (hihalf ? 8 : 0);
        #pragma unroll
        for (int c = 0; c < 4; ++c) {
            float4 lo0 = *(const float4*)(qb + c * 32);
            float4 lo1 = *(const float4*)(qb + c * 32 + 4);
            float4 hi0 = *(const float4*)(qb + c * 32 + 16);
            float4 hi1 = *(const float4*)(qb + c * 32 + 20);
            v16h a;
            a[0]=(_Float16)(lo0.x*qscale); a[1]=(_Float16)(lo0.y*qscale);
            a[2]=(_Float16)(lo0.z*qscale); a[3]=(_Float16)(lo0.w*qscale);
            a[4]=(_Float16)(lo1.x*qscale); a[5]=(_Float16)(lo1.y*qscale);
            a[6]=(_Float16)(lo1.z*qscale); a[7]=(_Float16)(lo1.w*qscale);
            a[8]=(_Float16)(hi0.x*qscale); a[9]=(_Float16)(hi0.y*qscale);
            a[10]=(_Float16)(hi0.z*qscale); a[11]=(_Float16)(hi0.w*qscale);
            a[12]=(_Float16)(hi1.x*qscale); a[13]=(_Float16)(hi1.y*qscale);
            a[14]=(_Float16)(hi1.z*qscale); a[15]=(_Float16)(hi1.w*qscale);
            aq[c] = a;
        }
    }

    v8f ctx[8];
    #pragma unroll
    for (int n = 0; n < 8; ++n) ctx[n] = (v8f){};
    float mrun[8], lrun[8];
    #pragma unroll
    for (int r = 0; r < 8; ++r) { mrun[r] = -3.0e38f; lrun[r] = 0.0f; }

    for (int kc = 0; kc < NCHUNK; ++kc) {
        const int k0 = kc * BK;
        #pragma unroll
        for (int i = 0; i < (BK * DH) / (128 * 4); ++i) {
            int idx = t + i * 128;
            int row = idx >> 5;
            int d4  = (idx & 31) * 4;
            const float* gp = K + ((size_t)(k0 + row) * NH + h) * DH + d4;
            float4 kv = *(const float4*)gp;
            v4h kh = { (_Float16)kv.x, (_Float16)kv.y, (_Float16)kv.z, (_Float16)kv.w };
            *(v4h*)&Klds[row][d4] = kh;
            if (kc + 1 < NCHUNK) __builtin_prefetch(gp + (size_t)BK * NH * DH, 0, 0);
        }
        {
            const int d = t;
            #pragma unroll
            for (int g = 0; g < BK / 4; ++g) {
                size_t base = ((size_t)(k0 + g * 4) * NH + h) * DH + d;
                float a0 = V[base];
                float a1 = V[base + (size_t)NH * DH];
                float a2 = V[base + (size_t)2 * NH * DH];
                float a3 = V[base + (size_t)3 * NH * DH];
                v4h vh = { (_Float16)a0, (_Float16)a1, (_Float16)a2, (_Float16)a3 };
                *(v4h*)&Vlds[d][g * 4] = vh;
                if (kc + 1 < NCHUNK)
                    __builtin_prefetch(&V[base + (size_t)BK * NH * DH], 0, 0);
            }
        }
        __syncthreads();

        v8f s[4];
        #pragma unroll
        for (int kt = 0; kt < 4; ++kt) s[kt] = (v8f){};
        #pragma unroll
        for (int c = 0; c < 4; ++c) {
            #pragma unroll
            for (int kt = 0; kt < 4; ++kt) {
                const _Float16* kp = &Klds[kt * 16 + l16][c * 32 + (hihalf ? 8 : 0)];
                v16h kb = join8(*(const v8h*)kp, *(const v8h*)(kp + 16));
                s[kt] = __builtin_amdgcn_wmma_f32_16x16x32_f16(false, aq[c], false, kb,
                                                               (short)0, s[kt], false, false);
            }
        }
        #pragma unroll
        for (int r = 0; r < 8; ++r) {
            float mx = fmaxf(fmaxf(s[0][r], s[1][r]), fmaxf(s[2][r], s[3][r]));
            mx = row16_max(mx);
            float mo = mrun[r];
            float mn = fmaxf(mo, mx);
            float sc = __builtin_amdgcn_exp2f(mo - mn);
            mrun[r]  = mn;
            float e0 = __builtin_amdgcn_exp2f(s[0][r] - mn);
            float e1 = __builtin_amdgcn_exp2f(s[1][r] - mn);
            float e2 = __builtin_amdgcn_exp2f(s[2][r] - mn);
            float e3 = __builtin_amdgcn_exp2f(s[3][r] - mn);
            s[0][r] = e0; s[1][r] = e1; s[2][r] = e2; s[3][r] = e3;
            float rs = row16_sum((e0 + e1) + (e2 + e3));
            lrun[r] = lrun[r] * sc + rs;
            #pragma unroll
            for (int n = 0; n < 8; ++n) ctx[n][r] *= sc;
        }
        {
            const int rbase = hihalf ? 8 : 0;
            #pragma unroll
            for (int r = 0; r < 8; ++r) {
                #pragma unroll
                for (int kt = 0; kt < 4; ++kt)
                    Plds[w][rbase + r][l16 + kt * 16] = (_Float16)s[kt][r];
            }
        }
        const _Float16* pp = &Plds[w][l16][hihalf ? 16 : 0];
        v16h pfrag0 = join8(*(const v8h*)pp,        *(const v8h*)(pp + 8));
        v16h pfrag1 = join8(*(const v8h*)(pp + 32), *(const v8h*)(pp + 40));
        #pragma unroll
        for (int n = 0; n < 8; ++n) {
            const _Float16* vp = &Vlds[n * 16 + l16][hihalf ? 16 : 0];
            v16h vb0 = join8(*(const v8h*)vp,        *(const v8h*)(vp + 8));
            ctx[n] = __builtin_amdgcn_wmma_f32_16x16x32_f16(false, pfrag0, false, vb0,
                                                            (short)0, ctx[n], false, false);
            v16h vb1 = join8(*(const v8h*)(vp + 32), *(const v8h*)(vp + 40));
            ctx[n] = __builtin_amdgcn_wmma_f32_16x16x32_f16(false, pfrag1, false, vb1,
                                                            (short)0, ctx[n], false, false);
        }
        __syncthreads();
    }

    #pragma unroll
    for (int r = 0; r < 8; ++r) {
        float inv = 1.0f / lrun[r];
        int row = qrow0 + (hihalf ? 8 : 0) + r;
        float* ob = O + ((size_t)row * NH + h) * DH + l16;
        #pragma unroll
        for (int n = 0; n < 8; ++n) ob[n * 16] = ctx[n][r] * inv;
    }
}

extern "C" void kernel_launch(void* const* d_in, const int* in_sizes, int n_in,
                              void* d_out, int out_size, void* d_ws, size_t ws_size,
                              hipStream_t stream) {
    (void)in_sizes; (void)n_in; (void)out_size;
    const float* q = (const float*)d_in[0];
    const float* k = (const float*)d_in[1];
    const float* v = (const float*)d_in[2];
    // d_in[3..7]: permutation indices cancel exactly (col-perm is softmax
    // invariant; head/row perms are undone by their deperms) -> unused.
    float* out = (float*)d_out;

    const size_t elems = (size_t)NH * S_LEN * DH;
    const size_t need  = elems * 2 /*f16*/ * 2 /*K + V^T*/;
    dim3 grid(S_LEN / BQ, NH);

    if (d_ws && ws_size >= need) {
        _Float16* K16  = (_Float16*)d_ws;
        _Float16* V16T = K16 + elems;
        cvt_k_kernel<<<(int)(elems / 256), 256, 0, stream>>>(k, K16);
        cvt_vt_kernel<<<dim3(S_LEN / 64, DH / 64, NH), 256, 0, stream>>>(v, V16T);
        fa_f16wmma_tdm_kernel<<<grid, dim3(128), 0, stream>>>(q, K16, V16T, out);
    } else {
        fa_f16wmma_kernel<<<grid, dim3(128), 0, stream>>>(q, k, v, out);
    }
}
